// TransformerEncoderBlock_76605036691503
// MI455X (gfx1250) — matrix operand order repaired and checked by reading, not compiled, not run on hardware
//
#include <hip/hip_runtime.h>
#include <hip/hip_bf16.h>
#include <cmath>

// ---------------- types / constants ----------------
typedef __attribute__((ext_vector_type(16))) _Float16 v16h;
typedef __attribute__((ext_vector_type(8)))  _Float16 v8h;
typedef __attribute__((ext_vector_type(8)))  float    v8f;
typedef __attribute__((ext_vector_type(4)))  unsigned int u32x4;
typedef __attribute__((ext_vector_type(8)))  int          i32x8;
typedef __attribute__((ext_vector_type(4)))  int          i32x4;

#define D_MODEL 1024
#define N_HEADS 16
#define D_KH    64
#define D_FF    4096
#define SEQ     2048
#define BATCH   4
#define M_ROWS  (BATCH * SEQ)   // 8192 tokens

#if defined(__has_builtin)
#if __has_builtin(__builtin_amdgcn_tensor_load_to_lds) && \
    __has_builtin(__builtin_amdgcn_s_wait_tensorcnt)
#define USE_TDM 1
#endif
#endif

__device__ __forceinline__ v8f wmma16(v16h a, v16h b, v8f c) {
  // D = A(16x32 f16) * B(32x16 f16) + C(16x16 f32)
  return __builtin_amdgcn_wmma_f32_16x16x32_f16(false, a, false, b, (short)0, c,
                                                false, false);
}

union V16 { v16h v; v8h h2[2]; _Float16 e[16]; };

// ---------------- weight convert + transpose: W[K][N] f32 -> WT[N][K] f16 ----
__global__ __launch_bounds__(256)
void k_wconv(const float* __restrict__ W, _Float16* __restrict__ WT,
             int K, int N) {
  size_t tid = (size_t)blockIdx.x * 256 + threadIdx.x;
  size_t total = (size_t)K * N;
  if (tid >= total) return;
  size_t n = tid / K, k = tid % K;
  WT[tid] = (_Float16)W[k * (size_t)N + n];
}

// ---------------- layernorm (row of 1024) -> f16 ----------------
__device__ __forceinline__ float blockReduceSum256(float s, volatile float* sh) {
#pragma unroll
  for (int o = 16; o > 0; o >>= 1) s += __shfl_xor(s, o, 32);
  if ((threadIdx.x & 31) == 0) sh[threadIdx.x >> 5] = s;
  __syncthreads();
  float tot = 0.f;
#pragma unroll
  for (int i = 0; i < 8; ++i) tot += sh[i];
  return tot;
}

__global__ __launch_bounds__(256)
void k_layernorm(const float* __restrict__ x, const float* __restrict__ g,
                 const float* __restrict__ bta, _Float16* __restrict__ out) {
  constexpr int D = D_MODEL;
  __shared__ float sred0[8];
  __shared__ float sred1[8];
  const int row = blockIdx.x;
  const float* xr = x + (size_t)row * D;
  float v[4];
  float s = 0.f;
#pragma unroll
  for (int j = 0; j < 4; ++j) { v[j] = xr[threadIdx.x + j * 256]; s += v[j]; }
  float mu = blockReduceSum256(s, sred0) * (1.0f / D);
  float vs = 0.f;
#pragma unroll
  for (int j = 0; j < 4; ++j) { float d = v[j] - mu; vs += d * d; }
  float var = blockReduceSum256(vs, sred1) * (1.0f / D);
  float rs = rsqrtf(var + 1e-5f);
  _Float16* orow = out + (size_t)row * D;
#pragma unroll
  for (int j = 0; j < 4; ++j) {
    int i = threadIdx.x + j * 256;
    orow[i] = (_Float16)((v[j] - mu) * rs * g[i] + bta[i]);
  }
}

// ---------------- WMMA GEMM, 128x128 macro-tile, LDS-staged B ----------------
// A: f16 row-major [M,K]. BT: f16 [N,K]. 256 threads = 8 waves; wave w owns
// rows [blk.x*128 + w*16, +16), all waves share the B tile staged in LDS.
enum { EPI_F16 = 0, EPI_VT = 1, EPI_GELU = 2, EPI_RES = 3 };

template <int EPI>
__global__ __launch_bounds__(256)
void k_gemm(const _Float16* __restrict__ A, const _Float16* __restrict__ BT,
            const float* __restrict__ bias, int M, int N, int K,
            void* __restrict__ out, const float* __restrict__ res) {
  __shared__ __align__(32) _Float16 ldsB[128 * 32];  // [n_local][k_local] 8 KB
  const int tid  = threadIdx.x;
  const int lane = tid & 31;
  const int wv   = tid >> 5;
  const int hx   = lane >> 4;
  const int l16  = lane & 15;
  const int mBase = blockIdx.x * 128 + wv * 16;
  const int nBase = blockIdx.y * 128;

  v8f acc[8];
#pragma unroll
  for (int t = 0; t < 8; ++t)
#pragma unroll
    for (int r = 0; r < 8; ++r) acc[t][r] = 0.f;

  const _Float16* arowb = A + (size_t)(mBase + l16) * K;

  for (int k0 = 0; k0 < K; k0 += 32) {
    __syncthreads();  // previous tile fully consumed
#if defined(USE_TDM)
    if (wv == 0) {
      // Tensor DMA: 2D tile, 128 rows x 64B (data_size=8B, tile_dim0=8)
      const unsigned long long ga =
          (unsigned long long)(BT + (size_t)nBase * K + k0);
      const unsigned ldsAddr = (unsigned)(unsigned long long)(void*)ldsB;
      const unsigned td0 = (unsigned)(K >> 2);   // tensor dim0 in 8B units
      const unsigned td1 = (unsigned)N;
      const unsigned long long st0 = (unsigned long long)(K >> 2);
      u32x4 g0;
      g0[0] = 1u;                                        // count=1
      g0[1] = ldsAddr;                                   // lds_addr
      g0[2] = (unsigned)ga;                              // global_addr lo
      g0[3] = (unsigned)((ga >> 32) & 0x1FFFFFFu) | (2u << 30);  // hi | type=2
      i32x8 g1;
      g1[0] = (3 << 16);                                 // data_size=8B
      g1[1] = (int)((td0 & 0xFFFFu) << 16);              // tensor_dim0 lo16
      g1[2] = (int)((td0 >> 16) & 0xFFFFu) | (int)((td1 & 0xFFFFu) << 16);
      g1[3] = (int)((td1 >> 16) & 0xFFFFu) | (8 << 16);  // tile_dim0=8
      g1[4] = 128;                                       // tile_dim1=128
      g1[5] = (int)(st0 & 0xFFFFFFFFull);                // dim0 stride lo
      g1[6] = (int)((st0 >> 32) & 0xFFFFull);            // stride hi | stride1=0
      g1[7] = 0;
      i32x4 z4;
      z4[0] = 0; z4[1] = 0; z4[2] = 0; z4[3] = 0;
#if __clang_major__ >= 23
      i32x8 z8;
#pragma unroll
      for (int i = 0; i < 8; ++i) z8[i] = 0;
      __builtin_amdgcn_tensor_load_to_lds(g0, g1, z4, z4, z8, 0);
#else
      __builtin_amdgcn_tensor_load_to_lds(g0, g1, z4, z4, 0);
#endif
      __builtin_amdgcn_s_wait_tensorcnt(0);
    }
#else
    {
      // cooperative staging: 2 threads per row, 32B each
      const int rr = tid >> 1;
      const int kk = (tid & 1) * 16;
      v16h tmp = *(const v16h*)(BT + (size_t)(nBase + rr) * K + k0 + kk);
      *(v16h*)&ldsB[rr * 32 + kk] = tmp;
    }
#endif
    __syncthreads();  // B tile visible to all waves

    // A fragment (contiguous k = 16*hx + s, same order as the B tile)
    V16 af;
    af.h2[0] = *(const v8h*)(arowb + k0 + hx * 16);
    af.h2[1] = *(const v8h*)(arowb + k0 + hx * 16 + 8);
    __builtin_prefetch(arowb + k0 + 32, 0, 1);
#pragma unroll
    for (int t = 0; t < 8; ++t) {
      v16h bf = *(const v16h*)&ldsB[(t * 16 + l16) * 32 + hx * 16];
      acc[t] = wmma16(af.v, bf, acc[t]);
    }
  }

  // epilogue: element (row mBase + r + 8*hx, col nBase + t*16 + l16)
#pragma unroll
  for (int t = 0; t < 8; ++t) {
    const int n = nBase + t * 16 + l16;
    const float bn = bias[n];
#pragma unroll
    for (int r = 0; r < 8; ++r) {
      const int m = mBase + r + 8 * hx;
      float val = acc[t][r] + bn;
      if constexpr (EPI == EPI_F16) {
        ((_Float16*)out)[(size_t)m * N + n] = (_Float16)val;
      } else if constexpr (EPI == EPI_VT) {
        int b = m >> 11, s = m & (SEQ - 1);
        int hh = n >> 6, d = n & 63;
        ((_Float16*)out)[(((size_t)b * N_HEADS + hh) * D_KH + d) * SEQ + s] =
            (_Float16)val;
      } else if constexpr (EPI == EPI_GELU) {
        float gv = 0.5f * val * (1.0f + erff(val * 0.70710678118654752f));
        ((_Float16*)out)[(size_t)m * N + n] = (_Float16)gv;
      } else {  // EPI_RES
        ((float*)out)[(size_t)m * N + n] = res[(size_t)m * N + n] + val;
      }
    }
  }
}

// ---------------- flash attention (transposed scores) ----------------------
// One wave per (b, h, 16 q-rows). S^T = K_tile(16x64) @ Q^T(64x16): in the
// C-layout the q index is the lane index, so softmax stats are per-lane.
__global__ __launch_bounds__(32)
void k_attention(const _Float16* __restrict__ Q, const _Float16* __restrict__ Km,
                 const _Float16* __restrict__ VT, const int* __restrict__ mask,
                 _Float16* __restrict__ out) {
  const int lane = threadIdx.x;
  const int hx   = lane >> 4;
  const int l16  = lane & 15;
  const int qt = blockIdx.x;
  const int h  = blockIdx.y;
  const int b  = blockIdx.z;
  const int qBase = qt * 16;

  // Q as B-fragments of Q^T (col q = l16, k-dim split by half); fold 1/sqrt(Dk)
  const _Float16* qrow =
      Q + ((size_t)b * SEQ + qBase + l16) * D_MODEL + h * D_KH + hx * 16;
  V16 qb0, qb1;
  qb0.v = *(const v16h*)(qrow);
  qb1.v = *(const v16h*)(qrow + 32);
#pragma unroll
  for (int i = 0; i < 16; ++i) {
    qb0.e[i] = (_Float16)((float)qb0.e[i] * 0.125f);
    qb1.e[i] = (_Float16)((float)qb1.e[i] * 0.125f);
  }

  float m_l = -INFINITY, l_l = 0.f;  // per-lane stats for q = qBase + l16
  v8f acc[4];                        // ctx^T: acc[t][r] = (d=t*16+r+8hx, q=l16)
#pragma unroll
  for (int t = 0; t < 4; ++t)
#pragma unroll
    for (int r = 0; r < 8; ++r) acc[t][r] = 0.f;

  const int* maskb = mask + (size_t)b * SEQ;
  const size_t vtOff = ((size_t)b * N_HEADS + h) * D_KH * SEQ;

  for (int kt = 0; kt < SEQ / 16; ++kt) {
    const int kBase = kt * 16;
    // K tile as A-fragments (row = key = kBase + l16)
    const _Float16* krow =
        Km + ((size_t)b * SEQ + kBase + l16) * D_MODEL + h * D_KH;
    V16 ka0, ka1;
    ka0.h2[0] = *(const v8h*)(krow + hx * 16);
    ka0.h2[1] = *(const v8h*)(krow + hx * 16 + 8);
    ka1.h2[0] = *(const v8h*)(krow + 32 + hx * 16);
    ka1.h2[1] = *(const v8h*)(krow + 32 + hx * 16 + 8);
    v8f s;
#pragma unroll
    for (int r = 0; r < 8; ++r) s[r] = 0.f;
    s = wmma16(ka0.v, qb0.v, s);
    s = wmma16(ka1.v, qb1.v, s);
    // s[r] = S^T(key = kBase + r + 8*hx, q = qBase + l16), already scaled

    // mask per key-row (8 contiguous ints for this half)
    const int* mrow = maskb + kBase + 8 * hx;
    int mk[8];
    *(int4*)&mk[0] = *(const int4*)(mrow);
    *(int4*)&mk[4] = *(const int4*)(mrow + 4);
    float sv[8];
#pragma unroll
    for (int r = 0; r < 8; ++r) sv[r] = (mk[r] != 0) ? s[r] : -INFINITY;

    // per-q (per-lane) max over 16 keys: 7 in-lane + 1 cross-half shuffle
    float mx = sv[0];
#pragma unroll
    for (int r = 1; r < 8; ++r) mx = fmaxf(mx, sv[r]);
    mx = fmaxf(mx, __shfl_xor(mx, 16, 32));
    const float mnew = fmaxf(m_l, mx);
    float p[8], ps = 0.f;
#pragma unroll
    for (int r = 0; r < 8; ++r) {
      p[r] = (sv[r] == -INFINITY) ? 0.f : __expf(sv[r] - mnew);
      ps += p[r];
    }
    ps += __shfl_xor(ps, 16, 32);
    const float sc = (m_l == -INFINITY) ? 0.f : __expf(m_l - mnew);
    l_l = l_l * sc + ps;
    m_l = mnew;
#pragma unroll
    for (int t = 0; t < 4; ++t)
#pragma unroll
      for (int r = 0; r < 8; ++r) acc[t][r] *= sc;

    // build P^T B-fragment (col q = l16): elements 0..7 carry keys 8*hx+0..7,
    // elements 8..15 are the K=16..31 zero padding.
    float po[8];
#pragma unroll
    for (int r = 0; r < 8; ++r) po[r] = __shfl_xor(p[r], 16, 32);
    V16 pb;
#pragma unroll
    for (int i = 0; i < 8; ++i) {
      pb.e[i]     = (_Float16)p[i];
      pb.e[i + 8] = (_Float16)0.f;
    }

    // ctx^T += V^T * P^T  (A-fragment of V^T contiguous over keys)
#pragma unroll
    for (int t = 0; t < 4; ++t) {
      V16 va;
      va.h2[0] =
          *(const v8h*)(VT + vtOff + (size_t)(t * 16 + l16) * SEQ + kBase + hx * 8);
#pragma unroll
      for (int i = 8; i < 16; ++i) va.e[i] = (_Float16)0.f;  // key pad 16..31
      acc[t] = wmma16(va.v, pb.v, acc[t]);
    }
  }

  // normalize + store: contiguous v8h per t (d = t*16 + 8*hx + r)
  const float inv = (l_l > 0.f) ? (1.0f / l_l) : 0.f;
  _Float16* orow = out + ((size_t)b * SEQ + qBase + l16) * D_MODEL + h * D_KH;
#pragma unroll
  for (int t = 0; t < 4; ++t) {
    v8h ov;
#pragma unroll
    for (int r = 0; r < 8; ++r) ov[r] = (_Float16)(acc[t][r] * inv);
    *(v8h*)(orow + t * 16 + 8 * hx) = ov;
  }
}

// ---------------- orchestration ----------------
extern "C" void kernel_launch(void* const* d_in, const int* in_sizes, int n_in,
                              void* d_out, int out_size, void* d_ws,
                              size_t ws_size, hipStream_t stream) {
  const float* x    = (const float*)d_in[0];
  const int*   mask = (const int*)d_in[1];
  const float* Wq = (const float*)d_in[2];  const float* bq = (const float*)d_in[3];
  const float* Wk = (const float*)d_in[4];  const float* bk = (const float*)d_in[5];
  const float* Wv = (const float*)d_in[6];  const float* bv = (const float*)d_in[7];
  const float* Wo = (const float*)d_in[8];  const float* bo = (const float*)d_in[9];
  const float* W1 = (const float*)d_in[10]; const float* b1 = (const float*)d_in[11];
  const float* W2 = (const float*)d_in[12]; const float* b2 = (const float*)d_in[13];
  const float* ln1g = (const float*)d_in[14]; const float* ln1b = (const float*)d_in[15];
  const float* ln2g = (const float*)d_in[16]; const float* ln2b = (const float*)d_in[17];
  float* out = (float*)d_out;
  char* ws = (char*)d_ws;

  const size_t MB = 1ull << 20;
  _Float16* WqT = (_Float16*)(ws + 0 * MB);     // 2 MB
  _Float16* WkT = (_Float16*)(ws + 2 * MB);     // 2 MB
  _Float16* WvT = (_Float16*)(ws + 4 * MB);     // 2 MB
  _Float16* WoT = (_Float16*)(ws + 6 * MB);     // 2 MB
  _Float16* W1T = (_Float16*)(ws + 8 * MB);     // 8 MB
  _Float16* W2T = (_Float16*)(ws + 16 * MB);    // 8 MB
  _Float16* h1  = (_Float16*)(ws + 24 * MB);    // 16 MB
  _Float16* Qf  = (_Float16*)(ws + 40 * MB);    // 16 MB
  _Float16* Kf  = (_Float16*)(ws + 56 * MB);    // 16 MB
  _Float16* VTf = (_Float16*)(ws + 72 * MB);    // 16 MB  [B,H,Dk,S]
  _Float16* ctx = (_Float16*)(ws + 88 * MB);    // 16 MB
  float*    x1  = (float*)(ws + 104 * MB);      // 32 MB
  _Float16* h2  = (_Float16*)(ws + 136 * MB);   // 16 MB
  _Float16* ffh = (_Float16*)(ws + 152 * MB);   // 64 MB (ends at 216 MB)

  // 1) weight convert + transpose to f16 [N][K]
  k_wconv<<<(D_MODEL * D_MODEL + 255) / 256, 256, 0, stream>>>(Wq, WqT, D_MODEL, D_MODEL);
  k_wconv<<<(D_MODEL * D_MODEL + 255) / 256, 256, 0, stream>>>(Wk, WkT, D_MODEL, D_MODEL);
  k_wconv<<<(D_MODEL * D_MODEL + 255) / 256, 256, 0, stream>>>(Wv, WvT, D_MODEL, D_MODEL);
  k_wconv<<<(D_MODEL * D_MODEL + 255) / 256, 256, 0, stream>>>(Wo, WoT, D_MODEL, D_MODEL);
  k_wconv<<<(D_MODEL * D_FF   + 255) / 256, 256, 0, stream>>>(W1, W1T, D_MODEL, D_FF);
  k_wconv<<<(D_FF   * D_MODEL + 255) / 256, 256, 0, stream>>>(W2, W2T, D_FF, D_MODEL);

  // 2) LN1 -> h1 (f16)
  k_layernorm<<<M_ROWS, 256, 0, stream>>>(x, ln1g, ln1b, h1);

  // 3) Q/K/V projections (V written transposed per-head)
  dim3 gP(M_ROWS / 128, D_MODEL / 128);
  k_gemm<EPI_F16><<<gP, 256, 0, stream>>>(h1, WqT, bq, M_ROWS, D_MODEL, D_MODEL, Qf, nullptr);
  k_gemm<EPI_F16><<<gP, 256, 0, stream>>>(h1, WkT, bk, M_ROWS, D_MODEL, D_MODEL, Kf, nullptr);
  k_gemm<EPI_VT ><<<gP, 256, 0, stream>>>(h1, WvT, bv, M_ROWS, D_MODEL, D_MODEL, VTf, nullptr);

  // 4) flash attention
  dim3 gA(SEQ / 16, N_HEADS, BATCH);
  k_attention<<<gA, 32, 0, stream>>>(Qf, Kf, VTf, mask, ctx);

  // 5) output projection + residual -> x1 (f32)
  k_gemm<EPI_RES><<<gP, 256, 0, stream>>>(ctx, WoT, bo, M_ROWS, D_MODEL, D_MODEL, x1, x);

  // 6) LN2 -> h2 (f16)
  k_layernorm<<<M_ROWS, 256, 0, stream>>>(x1, ln2g, ln2b, h2);

  // 7) FFN1 with exact GELU -> ffh (f16)
  dim3 gF1(M_ROWS / 128, D_FF / 128);
  k_gemm<EPI_GELU><<<gF1, 256, 0, stream>>>(h2, W1T, b1, M_ROWS, D_FF, D_MODEL, ffh, nullptr);

  // 8) FFN2 + residual -> d_out (f32)
  k_gemm<EPI_RES><<<gP, 256, 0, stream>>>(ffh, W2T, b2, M_ROWS, D_MODEL, D_FF, out, x1);
}
